// toy_46617575031313
// MI455X (gfx1250) — matrix operand order repaired and checked by reading, not compiled, not run on hardware
//
#include <hip/hip_runtime.h>
#include <hip/hip_bf16.h>

// ---------------------------------------------------------------------------
// Blockwise-dequant GEMM:  C[M,N] = A[M,K] x (W[N,K] * scale[N/128,K/128])^T
// M = 8192, N = 6144, K = 3072 (fp32 in/out), bf16 WMMA compute on gfx1250.
// 128x256 block tile, 64x64 wave tile (4x4 WMMA frags), double-buffered LDS,
// global->register prefetch, 1 barrier per K-tile.
// ---------------------------------------------------------------------------

typedef __attribute__((ext_vector_type(16))) __bf16 v16bf;
typedef __attribute__((ext_vector_type(8)))  __bf16 v8bf;
typedef __attribute__((ext_vector_type(8)))  float  v8f;

#define BM 128
#define BN 256
#define BK 32
#define PAD 8                    // 40-element rows: 80B stride, 16B aligned,
#define LDSTR (BK + PAD)         // rows 20 dwords apart -> full 64-bank spread

union FragBF16 {                 // v16bf = two contiguous v8bf halves
    v16bf v;
    v8bf  h[2];
};

__device__ __forceinline__ v8bf cvt8(float4 a, float4 b) {
    v8bf r;
    r[0] = (__bf16)a.x; r[1] = (__bf16)a.y; r[2] = (__bf16)a.z; r[3] = (__bf16)a.w;
    r[4] = (__bf16)b.x; r[5] = (__bf16)b.y; r[6] = (__bf16)b.z; r[7] = (__bf16)b.w;
    return r;
}
__device__ __forceinline__ v8bf cvt8s(float4 a, float4 b, float s) {
    v8bf r;
    r[0] = (__bf16)(a.x * s); r[1] = (__bf16)(a.y * s);
    r[2] = (__bf16)(a.z * s); r[3] = (__bf16)(a.w * s);
    r[4] = (__bf16)(b.x * s); r[5] = (__bf16)(b.y * s);
    r[6] = (__bf16)(b.z * s); r[7] = (__bf16)(b.w * s);
    return r;
}

__global__ __launch_bounds__(256)
void gemm_dq_bf16_wmma(const float* __restrict__ A,    // [M][K] activations
                       const float* __restrict__ W,    // [N][K] fp8-valued weights (fp32 storage)
                       const float* __restrict__ Sinv, // [N/128][K/128] block scales
                       float* __restrict__ C,          // [M][N]
                       int M, int N, int K)
{
    __shared__ __bf16 As[2][BM][LDSTR];   // 20,480 B
    __shared__ __bf16 Ws[2][BN][LDSTR];   // 40,960 B  (total 61,440 B)

    const int tid   = threadIdx.x;
    const int lane  = tid & 31;
    const int wid   = tid >> 5;        // 8 waves
    const int waveM = wid >> 2;        // 0..1 -> 64-row slab
    const int waveN = wid & 3;         // 0..3 -> 64-col slab
    const int sel   = lane >> 4;       // half-wave K select
    const int l16   = lane & 15;

    const int mBase = blockIdx.y * BM;
    const int nBase = blockIdx.x * BN;
    const int nbk   = K >> 7;          // scale columns (K/128)

    // loader mapping: 2 threads per row, 16 contiguous floats per thread
    const int lrow = tid >> 1;         // 0..127
    const int lcol = (tid & 1) << 4;   // 0 or 16

    const float* paBase  = A + (size_t)(mBase + lrow) * K + lcol;          // A rows 0..127
    const float* pwBase0 = W + (size_t)(nBase + lrow) * K + lcol;          // W rows 0..127
    const float* pwBase1 = W + (size_t)(nBase + 128 + lrow) * K + lcol;    // W rows 128..255
    const float* srow0   = Sinv + (size_t)(2 * blockIdx.x)     * nbk;      // scale row for n 0..127
    const float* srow1   = Sinv + (size_t)(2 * blockIdx.x + 1) * nbk;      // scale row for n 128..255

    float4 ra[4], rw0[4], rw1[4];      // in-flight staging registers

    v8f zero = {};
    v8f acc[4][4];
#pragma unroll
    for (int mi = 0; mi < 4; ++mi)
#pragma unroll
        for (int ni = 0; ni < 4; ++ni) acc[mi][ni] = zero;

    // ---- pipeline helpers (inlined) ----
    auto issue_loads = [&](int kt) {
        const float4* pa = (const float4*)(paBase  + kt * BK);
        const float4* p0 = (const float4*)(pwBase0 + kt * BK);
        const float4* p1 = (const float4*)(pwBase1 + kt * BK);
#pragma unroll
        for (int i = 0; i < 4; ++i) { ra[i] = pa[i]; rw0[i] = p0[i]; rw1[i] = p1[i]; }
    };

    auto stage = [&](int kt) {         // cvt fp32->bf16 (weights pre-scaled) -> LDS
        const int   kb = (kt * BK) >> 7;
        const float s0 = srow0[kb];
        const float s1 = srow1[kb];
        const int   b  = kt & 1;
        *(v8bf*)&As[b][lrow][lcol]           = cvt8 (ra[0],  ra[1]);
        *(v8bf*)&As[b][lrow][lcol + 8]       = cvt8 (ra[2],  ra[3]);
        *(v8bf*)&Ws[b][lrow][lcol]           = cvt8s(rw0[0], rw0[1], s0);
        *(v8bf*)&Ws[b][lrow][lcol + 8]       = cvt8s(rw0[2], rw0[3], s0);
        *(v8bf*)&Ws[b][128 + lrow][lcol]     = cvt8s(rw1[0], rw1[1], s1);
        *(v8bf*)&Ws[b][128 + lrow][lcol + 8] = cvt8s(rw1[2], rw1[3], s1);
    };

    auto compute = [&](int kt) {       // 16 x v_wmma_f32_16x16x32_bf16
        const int b = kt & 1;
        FragBF16 af[4];
#pragma unroll
        for (int mi = 0; mi < 4; ++mi) {
            // 16-bit A 16x32 layout: elems0-7 K=sel*8.., elems8-15 K=16+sel*8..
            const __bf16* p = &As[b][waveM * 64 + mi * 16 + l16][sel << 3];
            af[mi].h[0] = *(const v8bf*)p;
            af[mi].h[1] = *(const v8bf*)(p + 16);
        }
        FragBF16 bfr[4];
#pragma unroll
        for (int ni = 0; ni < 4; ++ni) {
            // 16-bit B 32x16 layout: elems0-7 K=sel*8.., elems8-15 K=16+sel*8..
            const __bf16* p = &Ws[b][waveN * 64 + ni * 16 + l16][sel << 3];
            bfr[ni].h[0] = *(const v8bf*)p;
            bfr[ni].h[1] = *(const v8bf*)(p + 16);
        }
#pragma unroll
        for (int mi = 0; mi < 4; ++mi)
#pragma unroll
            for (int ni = 0; ni < 4; ++ni)
                acc[mi][ni] = __builtin_amdgcn_wmma_f32_16x16x32_bf16(
                    false, af[mi].v, false, bfr[ni].v,
                    (short)0, acc[mi][ni], false, false);
    };

    // ---- 2-stage pipeline: prefetch kt+1 globals while computing kt ----
    const int T = K / BK;              // 96 K-tiles
    issue_loads(0);
    stage(0);
    __syncthreads();

    for (int kt = 0; kt < T - 1; ++kt) {
        issue_loads(kt + 1);           // in flight during compute
        compute(kt);                   // reads buf kt&1
        stage(kt + 1);                 // writes buf (kt+1)&1 (no reader conflict)
        __syncthreads();               // single barrier per K-tile
    }
    compute(T - 1);

    // ---- epilogue: C 16x16 f32 layout: VGPR r -> row sel*8+r, lane -> col ----
#pragma unroll
    for (int mi = 0; mi < 4; ++mi) {
        const int rowBase = mBase + waveM * 64 + mi * 16 + sel * 8;
#pragma unroll
        for (int ni = 0; ni < 4; ++ni) {
            const int col = nBase + waveN * 64 + ni * 16 + l16;
            float* cp = C + (size_t)rowBase * N + col;
#pragma unroll
            for (int r = 0; r < 8; ++r)
                cp[(size_t)r * N] = acc[mi][ni][r];
        }
    }
}

extern "C" void kernel_launch(void* const* d_in, const int* in_sizes, int n_in,
                              void* d_out, int out_size, void* d_ws, size_t ws_size,
                              hipStream_t stream) {
    const float* A    = (const float*)d_in[0];   // [B*S][K] fp32
    const float* W    = (const float*)d_in[1];   // [N][K]   fp32
    const float* Sinv = (const float*)d_in[2];   // [N/128][K/128] fp32
    float*       C    = (float*)d_out;           // [B*S][N] fp32

    const int K = 3072;
    const int N = 6144;
    const int M = in_sizes[0] / K;               // 8192

    dim3 grid(N / BN, M / BM);                   // 24 x 64
    gemm_dq_bf16_wmma<<<grid, 256, 0, stream>>>(A, W, Sinv, C, M, N, K);
}
